// PermutationEquivariantAttention_30709016167182
// MI455X (gfx1250) — matrix operand order repaired and checked by reading, not compiled, not run on hardware
//
#include <hip/hip_runtime.h>

#define D_MODEL 1024
#define N_HEADS 16
#define HEAD_DIM 64
#define SEQ 2048
#define BATCH 2
#define KSTEP 32

typedef __attribute__((ext_vector_type(8)))  float          v8f;
typedef __attribute__((ext_vector_type(16))) __bf16         v16bf;
typedef __attribute__((ext_vector_type(8)))  unsigned short ush8;

union BfPack { v16bf v; ush8 h[2]; };

__device__ inline v8f wmma_bf16(v16bf a, v16bf b, v8f c) {
  // (neg_a, A, neg_b, B, c_mod, C, reuse_a, reuse_b)
  return __builtin_amdgcn_wmma_f32_16x16x32_bf16(false, a, false, b, (short)0, c, false, false);
}

__device__ inline unsigned short f2bf(float f) {
  union { float f; unsigned u; } v; v.f = f;
  unsigned r = v.u + 0x7fffu + ((v.u >> 16) & 1u);
  return (unsigned short)(r >> 16);
}

__device__ inline v8f zero8() { v8f z = {0.f,0.f,0.f,0.f,0.f,0.f,0.f,0.f}; return z; }

// ---- CDNA5 async global->LDS (ASYNCcnt-tracked) ----
__device__ inline void async_copy_b128(unsigned lds_off, const void* gaddr) {
  asm volatile("global_load_async_to_lds_b128 %0, %1, off"
               :: "v"(lds_off), "v"(gaddr) : "memory");
}

__device__ inline void wait_async0() {
#if __has_builtin(__builtin_amdgcn_s_wait_asynccnt)
  __builtin_amdgcn_s_wait_asynccnt(0);
#else
  asm volatile("s_wait_asynccnt 0" ::: "memory");
#endif
}

// ---------------------------------------------------------------- convert
__global__ __launch_bounds__(256)
void cvt_bf16_kernel(const float* __restrict__ src, unsigned short* __restrict__ dst, int n) {
  for (int i = blockIdx.x * 256 + threadIdx.x; i < n; i += gridDim.x * 256)
    dst[i] = f2bf(src[i]);
}

__global__ __launch_bounds__(256)
void zero_kernel(float* __restrict__ p, int n) {
  for (int i = blockIdx.x * 256 + threadIdx.x; i < n; i += gridDim.x * 256)
    p[i] = 0.f;
}

// ---------------------------------------------------------------- GEMM
// C(M x 1024) = A(M x 1024 bf16) @ W^T (W is 1024x1024 bf16, row-major out x in) + bias
// B panel (64 cols x 32 k) staged once per block per K-step via async LDS copy,
// double buffered; A fragments register double-buffered; B fragments batched
// ahead of the WMMA burst so ds reads overlap matrix math.
// mode 0: f32 row-major to outF
// mode 1: bf16 head-split   out[((b*16+h)*SEQ + nr)*64 + j]
// mode 2: bf16 head-split-T out[((b*16+h)*64 + j)*SEQ + nr]
__global__ __launch_bounds__(256)
void gemm_bf16_kernel(const unsigned short* __restrict__ A,
                      const unsigned short* __restrict__ W,
                      const float* __restrict__ bias,
                      float* __restrict__ outF,
                      unsigned short* __restrict__ outB,
                      int mode) {
  __shared__ unsigned short Bpanel[2][64 * KSTEP];

  const int tid  = threadIdx.x;
  const int lane = tid & 31;
  const int w    = tid >> 5;
  const int hi   = lane >> 4;           // half-wave select
  const int hi8  = hi * 8;
  const int c0   = hi * 8;              // A K-chunk base
  const int bOff = hi * 8;              // B K-chunk base
  const int rowbase = blockIdx.x * 128 + w * 16;
  const int colbase = blockIdx.y * 64;

  // async staging: thread t -> col = t/4, k-chunk = (t%4)*8 ushorts (16B)
  const int scol   = tid >> 2;
  const int schunk = (tid & 3) * 8;
  const unsigned short* wsrc = W + (size_t)(colbase + scol) * D_MODEL + schunk;
  unsigned ldsdst[2];
  ldsdst[0] = (unsigned)(size_t)&Bpanel[0][scol * KSTEP + schunk];
  ldsdst[1] = (unsigned)(size_t)&Bpanel[1][scol * KSTEP + schunk];

  const unsigned short* arow = A + (size_t)(rowbase + (lane & 15)) * D_MODEL;

  v8f acc[4];
  #pragma unroll
  for (int t = 0; t < 4; ++t) acc[t] = zero8();

  // prologue: stage K-step 0
  async_copy_b128(ldsdst[0], wsrc);
  BfPack aCur, aNxt;
  aCur.h[0] = *(const ush8*)(arow + c0);
  aCur.h[1] = *(const ush8*)(arow + c0 + 16);
  wait_async0();
  __syncthreads();

  for (int kb = 0; kb < D_MODEL; kb += KSTEP) {
    const int buf = (kb >> 5) & 1;
    if (kb + KSTEP < D_MODEL) {
      // issue next panel into the other buffer (safe: its readers finished
      // before the barrier that ended the previous iteration)
      async_copy_b128(ldsdst[buf ^ 1], wsrc + kb + KSTEP);
      aNxt.h[0] = *(const ush8*)(arow + kb + KSTEP + c0);
      aNxt.h[1] = *(const ush8*)(arow + kb + KSTEP + c0 + 16);
    }
    // batch all B fragment reads, then the WMMA burst
    BfPack bfrag[4];
    #pragma unroll
    for (int t = 0; t < 4; ++t) {
      const unsigned short* lb = &Bpanel[buf][(t * 16 + (lane & 15)) * KSTEP + bOff];
      bfrag[t].h[0] = *(const ush8*)(lb);
      bfrag[t].h[1] = *(const ush8*)(lb + 16);
    }
    #pragma unroll
    for (int t = 0; t < 4; ++t)
      acc[t] = wmma_bf16(aCur.v, bfrag[t].v, acc[t]);
    aCur = aNxt;
    wait_async0();
    __syncthreads();
  }

  #pragma unroll
  for (int t = 0; t < 4; ++t) {
    #pragma unroll
    for (int r = 0; r < 8; ++r) {
      int row = rowbase + r + hi8;
      int col = colbase + t * 16 + (lane & 15);
      float val = acc[t][r] + bias[col];
      if (mode == 0) {
        outF[(size_t)row * D_MODEL + col] = val;
      } else {
        int bb = row >> 11, nr = row & (SEQ - 1);
        int h = col >> 6, j = col & 63;
        if (mode == 1)
          outB[(((size_t)(bb * N_HEADS + h)) * SEQ + nr) * HEAD_DIM + j] = f2bf(val);
        else
          outB[(((size_t)(bb * N_HEADS + h)) * HEAD_DIM + j) * SEQ + nr] = f2bf(val);
      }
    }
  }
}

// ---------------------------------------------------------------- attention
__device__ inline v8f score_tile(const unsigned short* __restrict__ khead, int keybase,
                                 int lane, int bOff, v16bf qa0, v16bf qa1) {
  const unsigned short* kr = khead + (size_t)(keybase + (lane & 15)) * HEAD_DIM;
  BfPack b0, b1;
  b0.h[0] = *(const ush8*)(kr + bOff);
  b0.h[1] = *(const ush8*)(kr + bOff + 16);
  b1.h[0] = *(const ush8*)(kr + 32 + bOff);
  b1.h[1] = *(const ush8*)(kr + 32 + bOff + 16);
  v8f acc = zero8();
  acc = wmma_bf16(qa0, b0.v, acc);
  acc = wmma_bf16(qa1, b1.v, acc);
  return acc;
}

__global__ __launch_bounds__(256)
void attn_kernel(const unsigned short* __restrict__ Qb,
                 const unsigned short* __restrict__ Kb,
                 const unsigned short* __restrict__ Vt,
                 const int*   __restrict__ perm,
                 const float* __restrict__ pbs,        // perm_bias_scale per head (16)
                 const float* __restrict__ ctemp,      // certainty_temp (1)
                 const float* __restrict__ cert_in,
                 unsigned short* __restrict__ attn_out, // (b, n, 1024) bf16
                 float* __restrict__ avg_attn,          // (b, n, n) f32, pre-zeroed
                 float* __restrict__ cert_out) {
  __shared__ float entLds[16];
  __shared__ unsigned short sc[8][16 * 32];

  const int tid  = threadIdx.x;
  const int lane = tid & 31;
  const int w    = tid >> 5;
  const int hi   = lane >> 4;
  const int hi8  = hi * 8;
  const int c0   = hi * 8;
  const int bOff = hi * 8;
  const int rowbase = blockIdx.x * 16;
  const int b       = blockIdx.y;

  if (tid < 16) entLds[tid] = 0.f;
  __syncthreads();

  // per-row permutation values (compile-time indexed, split by half-wave)
  float prL[8], prH[8];
  #pragma unroll
  for (int r = 0; r < 8; ++r) {
    prL[r] = (float)perm[b * SEQ + rowbase + r];
    prH[r] = (float)perm[b * SEQ + rowbase + 8 + r];
  }

  unsigned short* scp = sc[w];

  for (int hh = 0; hh < 2; ++hh) {
    const int h = w * 2 + hh;
    const float pscale = fabsf(pbs[h]);
    const unsigned short* qhead = Qb + ((size_t)(b * N_HEADS + h)) * SEQ * HEAD_DIM;
    const unsigned short* khead = Kb + ((size_t)(b * N_HEADS + h)) * SEQ * HEAD_DIM;
    const unsigned short* vhead = Vt + ((size_t)(b * N_HEADS + h)) * HEAD_DIM * SEQ;

    // A operand: Q rows for this 16-row block (2 K-tiles of 32 over hd=64)
    const unsigned short* qrow = qhead + (size_t)(rowbase + (lane & 15)) * HEAD_DIM;
    BfPack qa0, qa1;
    qa0.h[0] = *(const ush8*)(qrow + c0);
    qa0.h[1] = *(const ush8*)(qrow + c0 + 16);
    qa1.h[0] = *(const ush8*)(qrow + 32 + c0);
    qa1.h[1] = *(const ush8*)(qrow + 32 + c0 + 16);

    float mrow[8], lrow[8], trow[8];
    #pragma unroll
    for (int r = 0; r < 8; ++r) { mrow[r] = -3.0e38f; lrow[r] = 0.f; trow[r] = 0.f; }

    // ---- pass 1: online max / sum-exp / sum(e*s) ----
    for (int kt = 0; kt < SEQ / 16; ++kt) {
      v8f acc = score_tile(khead, kt * 16, lane, bOff, qa0.v, qa1.v);
      int key = kt * 16 + (lane & 15);
      float pk = (float)perm[b * SEQ + key];
      #pragma unroll
      for (int r = 0; r < 8; ++r) {
        float pr = hi ? prH[r] : prL[r];
        float s = acc[r] * 0.125f - fabsf(pr - pk) * pscale;
        float tmax = s;
        tmax = fmaxf(tmax, __shfl_xor(tmax, 1, 16));
        tmax = fmaxf(tmax, __shfl_xor(tmax, 2, 16));
        tmax = fmaxf(tmax, __shfl_xor(tmax, 4, 16));
        tmax = fmaxf(tmax, __shfl_xor(tmax, 8, 16));
        float nm = fmaxf(mrow[r], tmax);
        float f  = __expf(mrow[r] - nm);
        float e  = __expf(s - nm);
        float se = e, ses = e * s;
        se  += __shfl_xor(se, 1, 16);  ses += __shfl_xor(ses, 1, 16);
        se  += __shfl_xor(se, 2, 16);  ses += __shfl_xor(ses, 2, 16);
        se  += __shfl_xor(se, 4, 16);  ses += __shfl_xor(ses, 4, 16);
        se  += __shfl_xor(se, 8, 16);  ses += __shfl_xor(ses, 8, 16);
        lrow[r] = lrow[r] * f + se;
        trow[r] = trow[r] * f + ses;
        mrow[r] = nm;
      }
    }

    // entropy per row: H = m + log(l) - t/l  (one lane per half-wave adds)
    float linv[8];
    #pragma unroll
    for (int r = 0; r < 8; ++r) {
      linv[r] = 1.0f / lrow[r];
      float ent = mrow[r] + __logf(lrow[r]) - trow[r] * linv[r];
      if ((lane & 15) == 0) atomicAdd(&entLds[r + hi8], ent);
    }

    // ---- pass 2: probabilities, mean-attn accumulation, O = P @ V ----
    v8f o[4];
    #pragma unroll
    for (int j = 0; j < 4; ++j) o[j] = zero8();

    for (int ktp = 0; ktp < SEQ / 32; ++ktp) {
      #pragma unroll
      for (int half = 0; half < 2; ++half) {
        int kt = ktp * 2 + half;
        v8f acc = score_tile(khead, kt * 16, lane, bOff, qa0.v, qa1.v);
        int key = kt * 16 + (lane & 15);
        float pk = (float)perm[b * SEQ + key];
        #pragma unroll
        for (int r = 0; r < 8; ++r) {
          float pr = hi ? prH[r] : prL[r];
          float s = acc[r] * 0.125f - fabsf(pr - pk) * pscale;
          float e = __expf(s - mrow[r]);           // unnormalized prob
          // mean over heads of softmax prob: add e/l/16
          atomicAdd(avg_attn + ((size_t)(b * SEQ + rowbase + r + hi8)) * SEQ + key,
                    e * linv[r] * (1.0f / 16.0f));
          // D-layout -> LDS (row, local key) for A-layout reread
          scp[(r + hi8) * 32 + half * 16 + (lane & 15)] = f2bf(e);
        }
      }
      // reread as WMMA A operand (16 rows x 32 keys)
      const unsigned short* rp = scp + (lane & 15) * 32;
      BfPack pa;
      pa.h[0] = *(const ush8*)(rp + c0);
      pa.h[1] = *(const ush8*)(rp + c0 + 16);
      // batch all V fragment loads, then the WMMA burst
      BfPack vb[4];
      #pragma unroll
      for (int j = 0; j < 4; ++j) {
        const unsigned short* vr = vhead + (size_t)(j * 16 + (lane & 15)) * SEQ + ktp * 32 + bOff;
        vb[j].h[0] = *(const ush8*)(vr);
        vb[j].h[1] = *(const ush8*)(vr + 16);
      }
      #pragma unroll
      for (int j = 0; j < 4; ++j)
        o[j] = wmma_bf16(pa.v, vb[j].v, o[j]);
    }

    // store O / l as bf16 into (b, n, d) for the Wo GEMM
    #pragma unroll
    for (int j = 0; j < 4; ++j) {
      #pragma unroll
      for (int r = 0; r < 8; ++r) {
        size_t idx = ((size_t)(b * SEQ + rowbase + r + hi8)) * D_MODEL
                   + h * HEAD_DIM + j * 16 + (lane & 15);
        attn_out[idx] = f2bf(o[j][r] * linv[r]);
      }
    }
  }

  __syncthreads();
  if (tid < 16) {
    float entm = entLds[tid] * (1.0f / 16.0f);
    float z = ctemp[0] * (__logf((float)SEQ) - entm);
    float upd = 1.0f / (1.0f + __expf(-z));
    int gi = b * SEQ + rowbase + tid;
    cert_out[gi] = fmaxf(cert_in[gi], upd);
  }
}

// ---------------------------------------------------------------- Sinkhorn
__global__ __launch_bounds__(256)
void sink_init_kernel(float* __restrict__ P, int n) {
  for (int i = blockIdx.x * 256 + threadIdx.x; i < n; i += gridDim.x * 256) {
    float t = P[i] + 1e-10f;                 // exp(log(a+eps)/0.1) == (a+eps)^10
    float t2 = t * t, t4 = t2 * t2, t8 = t4 * t4;
    P[i] = t8 * t2;
  }
}

__global__ __launch_bounds__(256)
void rownorm_kernel(float* __restrict__ P) {
  __shared__ float red[256];
  float* rp = P + (size_t)blockIdx.x * SEQ;
  float s = 0.f;
  for (int c = threadIdx.x; c < SEQ; c += 256) s += rp[c];
  red[threadIdx.x] = s;
  __syncthreads();
  for (int st = 128; st > 0; st >>= 1) {
    if (threadIdx.x < st) red[threadIdx.x] += red[threadIdx.x + st];
    __syncthreads();
  }
  float inv = 1.0f / (red[0] + 1e-10f);
  for (int c = threadIdx.x; c < SEQ; c += 256) rp[c] *= inv;
}

__global__ __launch_bounds__(256)
void colnorm_kernel(float* __restrict__ P) {
  int col = blockIdx.x * 256 + threadIdx.x;
  float* bp = P + (size_t)blockIdx.y * SEQ * SEQ;
  float s = 0.f;
  for (int r = 0; r < SEQ; ++r) s += bp[(size_t)r * SEQ + col];
  float inv = 1.0f / (s + 1e-10f);
  for (int r = 0; r < SEQ; ++r) bp[(size_t)r * SEQ + col] *= inv;
}

__global__ __launch_bounds__(256)
void argmax_kernel(const float* __restrict__ P, const int* __restrict__ perm,
                   float* __restrict__ perm_out) {
  __shared__ float bvv[256];
  __shared__ int   bii[256];
  const float* rp = P + (size_t)blockIdx.x * SEQ;
  float best = -3.0e38f; int bidx = 0;
  for (int c = threadIdx.x; c < SEQ; c += 256) {
    float v = rp[c];
    if (v > best || (v == best && c < bidx)) { best = v; bidx = c; }
  }
  bvv[threadIdx.x] = best; bii[threadIdx.x] = bidx;
  __syncthreads();
  for (int st = 128; st > 0; st >>= 1) {
    if (threadIdx.x < st) {
      float ov = bvv[threadIdx.x + st]; int oi = bii[threadIdx.x + st];
      if (ov > bvv[threadIdx.x] || (ov == bvv[threadIdx.x] && oi < bii[threadIdx.x])) {
        bvv[threadIdx.x] = ov; bii[threadIdx.x] = oi;
      }
    }
    __syncthreads();
  }
  if (threadIdx.x == 0) {
    int b = blockIdx.x >> 11;
    perm_out[blockIdx.x] = (float)perm[b * SEQ + bii[0]];
  }
}

// ---------------------------------------------------------------- launch
extern "C" void kernel_launch(void* const* d_in, const int* in_sizes, int n_in,
                              void* d_out, int out_size, void* d_ws, size_t ws_size,
                              hipStream_t stream) {
  (void)in_sizes; (void)n_in; (void)out_size; (void)ws_size;

  const float* x      = (const float*)d_in[0];
  const float* cert   = (const float*)d_in[1];
  const int*   perm   = (const int*)  d_in[2];
  const float* Wq     = (const float*)d_in[3];
  const float* bq     = (const float*)d_in[4];
  const float* Wk     = (const float*)d_in[5];
  const float* bk     = (const float*)d_in[6];
  const float* Wv     = (const float*)d_in[7];
  const float* bv     = (const float*)d_in[8];
  const float* Wo     = (const float*)d_in[9];
  const float* bo     = (const float*)d_in[10];
  const float* pbs    = (const float*)d_in[11];
  const float* ctemp  = (const float*)d_in[12];

  float* out      = (float*)d_out;                                  // (b,n,d)
  float* cert_out = out + (size_t)BATCH * SEQ * D_MODEL;            // (b,n)
  float* perm_out = cert_out + (size_t)BATCH * SEQ;                 // (b,n)

  char* ws = (char*)d_ws;
  const size_t NX = (size_t)BATCH * SEQ * D_MODEL;                  // 4M
  const size_t NW = (size_t)D_MODEL * D_MODEL;                      // 1M
  unsigned short* xb   = (unsigned short*)(ws);                     // 8 MB
  unsigned short* wqb  = (unsigned short*)(ws + 8388608);
  unsigned short* wkb  = (unsigned short*)(ws + 10485760);
  unsigned short* wvb  = (unsigned short*)(ws + 12582912);
  unsigned short* wob  = (unsigned short*)(ws + 14680064);
  unsigned short* qb   = (unsigned short*)(ws + 16777216);          // (b,H,n,hd)
  unsigned short* kb_  = (unsigned short*)(ws + 25165824);          // (b,H,n,hd)
  unsigned short* vtb  = (unsigned short*)(ws + 33554432);          // (b,H,hd,n)
  unsigned short* aob  = (unsigned short*)(ws + 41943040);          // (b,n,d) bf16
  float*          avg  = (float*)        (ws + 50331648);           // (b,n,n) f32

  // 1) convert inputs to bf16
  cvt_bf16_kernel<<<2048, 256, 0, stream>>>(x,  xb,  (int)NX);
  cvt_bf16_kernel<<<1024, 256, 0, stream>>>(Wq, wqb, (int)NW);
  cvt_bf16_kernel<<<1024, 256, 0, stream>>>(Wk, wkb, (int)NW);
  cvt_bf16_kernel<<<1024, 256, 0, stream>>>(Wv, wvb, (int)NW);
  cvt_bf16_kernel<<<1024, 256, 0, stream>>>(Wo, wob, (int)NW);

  // 2) QKV projections (WMMA, async-LDS staged B)
  dim3 ggrid(BATCH * SEQ / 128, D_MODEL / 64);
  gemm_bf16_kernel<<<ggrid, 256, 0, stream>>>(xb, wqb, bq, nullptr, qb,  1);
  gemm_bf16_kernel<<<ggrid, 256, 0, stream>>>(xb, wkb, bk, nullptr, kb_, 1);
  gemm_bf16_kernel<<<ggrid, 256, 0, stream>>>(xb, wvb, bv, nullptr, vtb, 2);

  // 3) attention (WMMA flash-style, entropy, mean-attn accumulation)
  zero_kernel<<<4096, 256, 0, stream>>>(avg, BATCH * SEQ * SEQ);
  attn_kernel<<<dim3(SEQ / 16, BATCH), 256, 0, stream>>>(
      qb, kb_, vtb, perm, pbs, ctemp, cert, aob, avg, cert_out);

  // 4) output projection (WMMA) straight into d_out
  gemm_bf16_kernel<<<ggrid, 256, 0, stream>>>(aob, wob, bo, out, nullptr, 0);

  // 5) Sinkhorn + permutation update
  sink_init_kernel<<<4096, 256, 0, stream>>>(avg, BATCH * SEQ * SEQ);
  for (int it = 0; it < 10; ++it) {
    rownorm_kernel<<<BATCH * SEQ, 256, 0, stream>>>(avg);
    colnorm_kernel<<<dim3(SEQ / 256, BATCH), 256, 0, stream>>>(avg);
  }
  argmax_kernel<<<BATCH * SEQ, 256, 0, stream>>>(avg, perm, perm_out);
}
